// RecurrentBlock_76338748719962
// MI455X (gfx1250) — compile-verified, matrix operand order checked by reading, not run on hardware
//
#include <hip/hip_runtime.h>
#include <math.h>

typedef __attribute__((ext_vector_type(16))) __bf16 v16bf;
typedef __attribute__((ext_vector_type(8)))  float  v8f;

#define B_    16
#define TDEC  256
#define TXT   384
#define NMELD 80
#define PRED  256
#define VALD  512
#define ATTD  128
#define HID   1024
#define FILT  32
#define KSZ   31
#define OUTD  81

// ---------- helpers ----------

__device__ __forceinline__ unsigned short f2bf(float f) {
  unsigned int u = __builtin_bit_cast(unsigned int, f);
  if ((u & 0x7F800000u) != 0x7F800000u)          // keep inf/nan as-is
    u += 0x7FFFu + ((u >> 16) & 1u);             // round-to-nearest-even
  return (unsigned short)(u >> 16);
}

__device__ __forceinline__ float sigf(float x) { return 1.0f / (1.0f + __expf(-x)); }

// K index inside a 16x32 bf16 WMMA fragment for (lane, element e in v16bf).
// Layout per ISA 7.12.2: lanes 0-15 hold K 0..7 (VGPR0..3) and 16..23 (VGPR4..7),
// lanes 16-31 hold K 8..15 and 24..31; 2 bf16 per VGPR.
__device__ __forceinline__ int frag_k(int lane, int e) {
  int khalf = lane >> 4;
  return ((e >= 8) ? 16 : 0) + khalf * 8 + ((e >> 1) & 3) * 2 + (e & 1);
}

// ---------- setup kernels ----------

__global__ void zero_kernel(float* __restrict__ p, int n) {
  int i = blockIdx.x * 256 + threadIdx.x;
  if (i < n) p[i] = 0.0f;
}

// Pack [wih | whh] (rows = 4096 gate outputs, Ktot cols) into per-tile B fragments (bf16).
// dst index: (((nt*KT + kt)*32 + lane)*16 + e)
__global__ void pack_weights_kernel(const float* __restrict__ wih,
                                    const float* __restrict__ whh,
                                    int ksplit, int ktot, int KT,
                                    unsigned short* __restrict__ dst, int total) {
  int p = blockIdx.x * 256 + threadIdx.x;
  if (p >= total) return;
  int e    = p & 15;
  int lane = (p >> 4) & 31;
  int grp  = p >> 9;           // nt*KT + kt
  int kt   = grp % KT;
  int nt   = grp / KT;
  int n    = lane & 15;
  int j    = nt * 16 + n;
  int k    = kt * 32 + frag_k(lane, e);
  float v = (k < ksplit) ? wih[(size_t)j * ksplit + k]
                         : whh[(size_t)j * (ktot - ksplit) + (k - ksplit)];
  dst[p] = f2bf(v);
}

// prenet layer 1: x1 = relu(di @ w1^T), rows=B*T=4096, cols=256, K=80
__global__ void prenet1_kernel(const float* __restrict__ di,
                               const float* __restrict__ w1,
                               float* __restrict__ x1) {
  int idx = blockIdx.x * 256 + threadIdx.x;      // 4096*256
  int row = idx >> 8, col = idx & 255;
  const float* d = di + (size_t)row * NMELD;
  const float* w = w1 + (size_t)col * NMELD;
  float s = 0.f;
  for (int k = 0; k < NMELD; ++k) s += d[k] * w[k];
  x1[idx] = fmaxf(s, 0.f);
}

// prenet layer 2: x2 = relu(x1 @ w2^T), rows=4096, cols=256, K=256
__global__ void prenet2_kernel(const float* __restrict__ x1,
                               const float* __restrict__ w2,
                               float* __restrict__ x2) {
  int idx = blockIdx.x * 256 + threadIdx.x;
  int row = idx >> 8, col = idx & 255;
  const float* xr = x1 + (size_t)row * PRED;
  const float* w  = w2 + (size_t)col * PRED;
  float s = 0.f;
  for (int k = 0; k < PRED; ++k) s += xr[k] * w[k];
  x2[idx] = fmaxf(s, 0.f);
}

// key = value @ key_w^T + key_b : [b*384+t][a]
__global__ void key_kernel(const float* __restrict__ value,
                           const float* __restrict__ kw,
                           const float* __restrict__ kb,
                           float* __restrict__ keybuf) {
  int idx = blockIdx.x * 256 + threadIdx.x;      // 16*384*128
  int a  = idx & 127;
  int bt = idx >> 7;
  const float* v = value + (size_t)bt * VALD;
  const float* w = kw + (size_t)a * VALD;
  float s = kb[a];
  for (int k = 0; k < VALD; ++k) s += v[k] * w[k];
  keybuf[idx] = s;
}

// ---------- per-step kernels ----------

// Pack attention-LSTM input [prenet_t(256) | ctx(512) | h_a(1024)] into A fragments.
__global__ void pack_att_in_kernel(const float* __restrict__ pren,
                                   const float* __restrict__ ctx,
                                   const float* __restrict__ h_a,
                                   int t, unsigned short* __restrict__ dst) {
  int p = blockIdx.x * 256 + threadIdx.x;        // 16*1792 = 28672
  if (p >= 16 * 1792) return;
  int e    = p & 15;
  int lane = (p >> 4) & 31;
  int kt   = p >> 9;
  int m    = lane & 15;                          // batch
  int k    = kt * 32 + frag_k(lane, e);
  float v;
  if (k < PRED)            v = pren[((size_t)(m * TDEC + t)) * PRED + k];
  else if (k < PRED + VALD) v = ctx[m * VALD + (k - PRED)];
  else                      v = h_a[m * HID + (k - PRED - VALD)];
  dst[p] = f2bf(v);
}

// Pack decoder-LSTM input [h_a(1024) | ctx(512) | h_d(1024)] into A fragments.
__global__ void pack_dec_in_kernel(const float* __restrict__ h_a,
                                   const float* __restrict__ ctx,
                                   const float* __restrict__ h_d,
                                   unsigned short* __restrict__ dst) {
  int p = blockIdx.x * 256 + threadIdx.x;        // 16*2560 = 40960
  if (p >= 16 * 2560) return;
  int e    = p & 15;
  int lane = (p >> 4) & 31;
  int kt   = p >> 9;
  int m    = lane & 15;
  int k    = kt * 32 + frag_k(lane, e);
  float v;
  if (k < HID)            v = h_a[m * HID + k];
  else if (k < HID + VALD) v = ctx[m * VALD + (k - HID)];
  else                     v = h_d[m * HID + (k - HID - VALD)];
  dst[p] = f2bf(v);
}

// Fused LSTM gates (WMMA bf16) + cell nonlinearity.
// 64 blocks x 128 threads. Wave w (0..3) computes the 16x16 tile of gate chunk w
// (i/f/g/o) for hidden columns [blk*16, blk*16+16); LDS exchange, then cell update.
__global__ void lstm_gates_kernel(const unsigned short* __restrict__ Wp,
                                  const unsigned short* __restrict__ Ap,
                                  const float* __restrict__ bih,
                                  const float* __restrict__ bhh,
                                  float* __restrict__ h_st,
                                  float* __restrict__ c_st,
                                  int KT, float* __restrict__ feats, int t) {
  int tid  = threadIdx.x;
  int wave = tid >> 5;
  int lane = tid & 31;
  int blk  = blockIdx.x;
  int nt   = wave * 64 + blk;                    // tile over 4096 gate rows

  const v16bf* wfrag = (const v16bf*)Wp + (size_t)nt * KT * 32 + lane;
  const v16bf* afrag = (const v16bf*)Ap + lane;

  v8f acc = {};
  for (int kt = 0; kt < KT; ++kt) {
    v16bf a = afrag[kt * 32];
    v16bf b = wfrag[kt * 32];
    if (kt + 8 < KT)
      __builtin_prefetch((const void*)(wfrag + (size_t)(kt + 8) * 32), 0, 3);
    acc = __builtin_amdgcn_wmma_f32_16x16x32_bf16(false, a, false, b,
                                                  (short)0, acc, false, false);
  }

  __shared__ float g[4][16][16];
  int nn = lane & 15, mh = lane >> 4;
#pragma unroll
  for (int r = 0; r < 8; ++r) g[wave][r + 8 * mh][nn] = acc[r];
  __syncthreads();

  for (int idx = tid; idx < 256; idx += 128) {
    int m = idx >> 4, n = idx & 15;
    int j = blk * 16 + n;                        // hidden column 0..1023
    float gi = g[0][m][n] + bih[j]        + bhh[j];
    float gf = g[1][m][n] + bih[j + 1024] + bhh[j + 1024];
    float gg = g[2][m][n] + bih[j + 2048] + bhh[j + 2048];
    float go = g[3][m][n] + bih[j + 3072] + bhh[j + 3072];
    float cold = c_st[m * HID + j];
    float c2 = sigf(gf) * cold + sigf(gi) * tanhf(gg);
    float h  = sigf(go) * tanhf(c2);
    c_st[m * HID + j] = c2;
    h_st[m * HID + j] = h;
    if (feats) feats[((size_t)(m * TDEC + t)) * 1536 + j] = h;
  }
}

// q[b][a] = query_b[a] + h_a[b] . query_w[a]
__global__ void query_kernel(const float* __restrict__ h_a,
                             const float* __restrict__ qw,
                             const float* __restrict__ qb,
                             float* __restrict__ q) {
  int b = blockIdx.x, a = threadIdx.x;           // 16 x 128
  const float* hb = h_a + b * HID;
  const float* w  = qw + (size_t)a * HID;
  float s = qb[a];
  for (int k = 0; k < HID; ++k) s += hb[k] * w[k];
  q[b * ATTD + a] = s;
}

// Energy for one (t', b): location conv + dense + tanh + dot with v.
// grid (384, 16), 128 threads.
__global__ void att_energy_kernel(const float* __restrict__ awc,
                                  const float* __restrict__ q,
                                  const float* __restrict__ keybuf,
                                  const float* __restrict__ convw,
                                  const float* __restrict__ densew,
                                  const float* __restrict__ denseb,
                                  const float* __restrict__ vw,
                                  float* __restrict__ e) {
  int tpos = blockIdx.x;                         // 0..383
  int b    = blockIdx.y;                         // 0..15
  int tid  = threadIdx.x;                        // 0..127

  __shared__ float win[2][KSZ];
  if (tid < 2 * KSZ) {
    int c = tid / KSZ, k = tid % KSZ;
    int tt = tpos + k - (KSZ - 1) / 2;
    win[c][k] = (tt >= 0 && tt < TXT) ? awc[b * (2 * TXT) + c * TXT + tt] : 0.f;
  }
  __syncthreads();

  __shared__ float p[FILT];
  if (tid < FILT) {
    float s = 0.f;
    const float* w = convw + tid * (2 * KSZ);
    for (int k = 0; k < KSZ; ++k) s += win[0][k] * w[k] + win[1][k] * w[KSZ + k];
    p[tid] = s;
  }
  __syncthreads();

  int a = tid;
  float acc = denseb[a] + q[b * ATTD + a] + keybuf[((size_t)b * TXT + tpos) * ATTD + a];
  for (int f = 0; f < FILT; ++f) acc += densew[a * FILT + f] * p[f];
  float term = vw[a] * tanhf(acc);

  __shared__ float red[128];
  red[tid] = term;
  __syncthreads();
  for (int s = 64; s > 0; s >>= 1) {
    if (tid < s) red[tid] += red[tid + s];
    __syncthreads();
  }
  if (tid == 0) e[b * TXT + tpos] = red[0];
}

// Masked softmax over text dim, awc update, context, feats[...,1024:1536] write.
__global__ void att_softmax_ctx_kernel(const float* __restrict__ e,
                                       const int* __restrict__ vlen,
                                       const float* __restrict__ value,
                                       float* __restrict__ awc,
                                       float* __restrict__ ctx,
                                       float* __restrict__ feats, int t) {
  int b = blockIdx.x, tid = threadIdx.x;         // 16 x 256
  __shared__ float sal[TXT];
  __shared__ float red[256];
  int L = vlen[b];

  float mx = -INFINITY;
  for (int i = tid; i < TXT; i += 256) {
    float v = (i < L) ? e[b * TXT + i] : -INFINITY;
    sal[i] = v;
    mx = fmaxf(mx, v);
  }
  red[tid] = mx;
  __syncthreads();
  for (int s = 128; s > 0; s >>= 1) {
    if (tid < s) red[tid] = fmaxf(red[tid], red[tid + s]);
    __syncthreads();
  }
  float m = red[0];
  __syncthreads();

  float sum = 0.f;
  for (int i = tid; i < TXT; i += 256) {
    float ex = __expf(sal[i] - m);               // exp(-inf)=0 handles mask
    sal[i] = ex;
    sum += ex;
  }
  red[tid] = sum;
  __syncthreads();
  for (int s = 128; s > 0; s >>= 1) {
    if (tid < s) red[tid] += red[tid + s];
    __syncthreads();
  }
  float inv = 1.0f / red[0];
  __syncthreads();

  for (int i = tid; i < TXT; i += 256) {
    float a = sal[i] * inv;
    sal[i] = a;
    awc[b * (2 * TXT) + i] = a;                  // current
    awc[b * (2 * TXT) + TXT + i] += a;           // cumulative
  }
  __syncthreads();

  for (int d = tid; d < VALD; d += 256) {
    float s = 0.f;
    for (int i = 0; i < TXT; ++i) s += sal[i] * value[((size_t)b * TXT + i) * VALD + d];
    ctx[b * VALD + d] = s;
    feats[((size_t)(b * TDEC + t)) * 1536 + HID + d] = s;
  }
}

// out = feats @ proj_w^T + proj_b
__global__ void proj_kernel(const float* __restrict__ feats,
                            const float* __restrict__ pw,
                            const float* __restrict__ pb,
                            float* __restrict__ out) {
  int idx = blockIdx.x * 256 + threadIdx.x;      // 4096*81
  if (idx >= B_ * TDEC * OUTD) return;
  int row = idx / OUTD, o = idx % OUTD;
  const float* f = feats + (size_t)row * 1536;
  const float* w = pw + (size_t)o * 1536;
  float s = pb[o];
  for (int k = 0; k < 1536; ++k) s += f[k] * w[k];
  out[idx] = s;
}

// ---------- launch ----------

extern "C" void kernel_launch(void* const* d_in, const int* in_sizes, int n_in,
                              void* d_out, int out_size, void* d_ws, size_t ws_size,
                              hipStream_t stream) {
  const float* decoder_inputs = (const float*)d_in[0];
  const float* value       = (const float*)d_in[1];
  const int*   value_len   = (const int*)d_in[2];
  const float* prenet_w1   = (const float*)d_in[3];
  const float* prenet_w2   = (const float*)d_in[4];
  const float* key_w       = (const float*)d_in[5];
  const float* key_b       = (const float*)d_in[6];
  const float* query_w     = (const float*)d_in[7];
  const float* query_b     = (const float*)d_in[8];
  const float* loc_conv_w  = (const float*)d_in[9];
  const float* loc_dense_w = (const float*)d_in[10];
  const float* loc_dense_b = (const float*)d_in[11];
  const float* v_w         = (const float*)d_in[12];
  const float* att_wih     = (const float*)d_in[13];
  const float* att_whh     = (const float*)d_in[14];
  const float* att_bih     = (const float*)d_in[15];
  const float* att_bhh     = (const float*)d_in[16];
  const float* dec_wih     = (const float*)d_in[17];
  const float* dec_whh     = (const float*)d_in[18];
  const float* dec_bih     = (const float*)d_in[19];
  const float* dec_bhh     = (const float*)d_in[20];
  const float* proj_w      = (const float*)d_in[21];
  const float* proj_b      = (const float*)d_in[22];
  float* out = (float*)d_out;
  (void)in_sizes; (void)n_in; (void)out_size; (void)ws_size;

  char* ws = (char*)d_ws;
  size_t off = 0;
  auto alloc = [&](size_t bytes) -> char* {
    char* p = ws + off;
    off += (bytes + 255) & ~(size_t)255;
    return p;
  };

  size_t state0 = off;
  float* h_a = (float*)alloc((size_t)B_ * HID * 4);
  float* c_a = (float*)alloc((size_t)B_ * HID * 4);
  float* h_d = (float*)alloc((size_t)B_ * HID * 4);
  float* c_d = (float*)alloc((size_t)B_ * HID * 4);
  float* ctx = (float*)alloc((size_t)B_ * VALD * 4);
  float* awc = (float*)alloc((size_t)B_ * 2 * TXT * 4);
  size_t state1 = off;
  int stateFloats = (int)((state1 - state0) / 4);

  float* qbuf = (float*)alloc((size_t)B_ * ATTD * 4);
  float* ebuf = (float*)alloc((size_t)B_ * TXT * 4);
  unsigned short* pA_att = (unsigned short*)alloc((size_t)16 * 1792 * 2);
  unsigned short* pA_dec = (unsigned short*)alloc((size_t)16 * 2560 * 2);
  unsigned short* Wp_att = (unsigned short*)alloc((size_t)4096 * 1792 * 2);
  unsigned short* Wp_dec = (unsigned short*)alloc((size_t)4096 * 2560 * 2);
  float* x1    = (float*)alloc((size_t)4096 * PRED * 4);
  float* pren  = (float*)alloc((size_t)4096 * PRED * 4);
  float* keyb  = (float*)alloc((size_t)B_ * TXT * ATTD * 4);
  float* feats = (float*)alloc((size_t)4096 * 1536 * 4);

  zero_kernel<<<(stateFloats + 255) / 256, 256, 0, stream>>>((float*)(ws + state0), stateFloats);

  {
    int tot = 4096 * 1792;
    pack_weights_kernel<<<(tot + 255) / 256, 256, 0, stream>>>(
        att_wih, att_whh, 768, 1792, 56, Wp_att, tot);
  }
  {
    int tot = 4096 * 2560;
    pack_weights_kernel<<<(tot + 255) / 256, 256, 0, stream>>>(
        dec_wih, dec_whh, 1536, 2560, 80, Wp_dec, tot);
  }
  prenet1_kernel<<<4096, 256, 0, stream>>>(decoder_inputs, prenet_w1, x1);
  prenet2_kernel<<<4096, 256, 0, stream>>>(x1, prenet_w2, pren);
  key_kernel<<<(B_ * TXT * ATTD) / 256, 256, 0, stream>>>(value, key_w, key_b, keyb);

  for (int t = 0; t < TDEC; ++t) {
    pack_att_in_kernel<<<112, 256, 0, stream>>>(pren, ctx, h_a, t, pA_att);
    lstm_gates_kernel<<<64, 128, 0, stream>>>(Wp_att, pA_att, att_bih, att_bhh,
                                              h_a, c_a, 56, (float*)nullptr, t);
    query_kernel<<<16, 128, 0, stream>>>(h_a, query_w, query_b, qbuf);
    att_energy_kernel<<<dim3(TXT, B_), 128, 0, stream>>>(
        awc, qbuf, keyb, loc_conv_w, loc_dense_w, loc_dense_b, v_w, ebuf);
    att_softmax_ctx_kernel<<<B_, 256, 0, stream>>>(ebuf, value_len, value, awc,
                                                   ctx, feats, t);
    pack_dec_in_kernel<<<160, 256, 0, stream>>>(h_a, ctx, h_d, pA_dec);
    lstm_gates_kernel<<<64, 128, 0, stream>>>(Wp_dec, pA_dec, dec_bih, dec_bhh,
                                              h_d, c_d, 80, feats, t);
  }

  proj_kernel<<<(B_ * TDEC * OUTD + 255) / 256, 256, 0, stream>>>(feats, proj_w, proj_b, out);
}
